// TransformerEncoderLayer_modal_filter_v2_80857054314729
// MI455X (gfx1250) — compile-verified, matrix operand order checked by reading, not run on hardware
//
#include <hip/hip_runtime.h>
#include <hip/hip_bf16.h>
#include <stdint.h>

typedef __bf16 bf16;
typedef __attribute__((ext_vector_type(16))) __bf16 v16bf;
typedef __attribute__((ext_vector_type(8)))  float  v8f;

__device__ __forceinline__ v8f wmma_bf16(v16bf a, v16bf b, v8f c) {
  return __builtin_amdgcn_wmma_f32_16x16x32_bf16(false, a, false, b, (short)0, c, false, false);
}

// Fragment load: K runs [half8..half8+7] then [half8+16..half8+23], half8 = 8*(lane>=16).
__device__ __forceinline__ v16bf load_frag(const bf16* rowp, int half8) {
  union { v16bf v; uint4 q[2]; } u;
  u.q[0] = *(const uint4*)(rowp + half8);
  u.q[1] = *(const uint4*)(rowp + half8 + 16);
  return u.v;
}

// CDNA5 async global->LDS copy (16B per active lane), tracked by ASYNCcnt.
__device__ __forceinline__ void async_b128(unsigned lds, const void* g) {
  asm volatile("global_load_async_to_lds_b128 %0, %1, off" :: "v"(lds), "v"(g) : "memory");
}
__device__ __forceinline__ void wait_async0() {
  asm volatile("s_wait_asynccnt 0x0" ::: "memory");
}
__device__ __forceinline__ unsigned lds_addr(const void* p) {
  return (unsigned)(uintptr_t)p;  // LDS aperture: low 32 bits = LDS offset
}

__device__ __forceinline__ void pack16_store(const bf16* e16, bf16* dst) {
  union { bf16 e[16]; uint4 q[2]; } p;
#pragma unroll
  for (int i = 0; i < 16; ++i) p.e[i] = e16[i];
  ((uint4*)dst)[0] = p.q[0];
  ((uint4*)dst)[1] = p.q[1];
}

// ---------------------------------------------------------------------------
// f32 -> bf16 tensor conversion (one-time per tensor)
// ---------------------------------------------------------------------------
__global__ __launch_bounds__(256) void cvt_bf16_kernel(
    const float* __restrict__ in, bf16* __restrict__ out, int n)
{
  int i = (blockIdx.x * 256 + threadIdx.x) * 8;
  if (i >= n) return;
  float f[8];
  *(float4*)(f)     = *(const float4*)(in + i);
  *(float4*)(f + 4) = *(const float4*)(in + i + 4);
  union { bf16 e[8]; uint4 q; } p;
#pragma unroll
  for (int j = 0; j < 8; ++j) p.e[j] = (bf16)f[j];
  *(uint4*)(out + i) = p.q;
}

// ---------------------------------------------------------------------------
// Dense GEMM (bf16 in, f32 acc): Out = act(A[M,K] @ Wt[N,K]^T + bias + resid)
// 128x64 tile, 8 waves (4x2). LDS staged via async copies, double buffered.
// Requires M%128==0, N%64==0, K%32==0 (all launches satisfy this).
// ---------------------------------------------------------------------------
__global__ __launch_bounds__(256) void gemm_bf16_kernel(
    const bf16* __restrict__ A, const bf16* __restrict__ Wt,
    const float* __restrict__ bias, const float* __restrict__ resid,
    float* __restrict__ OutF, bf16* __restrict__ OutB,
    int M, int N, int K, int relu)
{
  __shared__ bf16 As[2][128][40];
  __shared__ bf16 Bs[2][64][40];
  constexpr unsigned ASZ = 128 * 40 * 2;  // bytes per A buffer
  constexpr unsigned BSZ = 64 * 40 * 2;
  const int tid  = threadIdx.x;
  const int lane = tid & 31, wave = tid >> 5;
  const int wrow = (wave >> 1) * 32, wcol = (wave & 1) * 32;
  const int tileM = blockIdx.y * 128, tileN = blockIdx.x * 64;
  const int half8 = (lane >> 4) << 3;
  const int l16   = lane & 15;

  // async staging map: 512 A-copies + 256 B-copies (16B each), 3 per thread
  const int arow0 = tid >> 2, aseg = tid & 3;
  const int arow1 = 64 + arow0;
  const bf16* agp0 = A + (size_t)(tileM + arow0) * K + aseg * 8;
  const bf16* agp1 = A + (size_t)(tileM + arow1) * K + aseg * 8;
  const bf16* bgp  = Wt + (size_t)(tileN + arow0) * K + aseg * 8;
  const unsigned asBase = lds_addr(&As[0][0][0]);
  const unsigned bsBase = lds_addr(&Bs[0][0][0]);
  const unsigned aoff0 = asBase + arow0 * 80 + aseg * 16;
  const unsigned aoff1 = asBase + arow1 * 80 + aseg * 16;
  const unsigned boff  = bsBase + arow0 * 80 + aseg * 16;

  auto stage = [&](int buf, int k0) {
    async_b128(aoff0 + buf * ASZ, agp0 + k0);
    async_b128(aoff1 + buf * ASZ, agp1 + k0);
    async_b128(boff  + buf * BSZ, bgp  + k0);
  };

  v8f acc[2][2] = {};
  const int KT = K >> 5;
  stage(0, 0);
  for (int kt = 0; kt < KT; ++kt) {
    wait_async0();
    __syncthreads();
    if (kt + 1 < KT) stage((kt + 1) & 1, (kt + 1) << 5);
    const int buf = kt & 1;
    v16bf af0 = load_frag(&As[buf][wrow + l16][0], half8);
    v16bf af1 = load_frag(&As[buf][wrow + 16 + l16][0], half8);
    v16bf bf0 = load_frag(&Bs[buf][wcol + l16][0], half8);
    v16bf bf1 = load_frag(&Bs[buf][wcol + 16 + l16][0], half8);
    acc[0][0] = wmma_bf16(af0, bf0, acc[0][0]);
    acc[0][1] = wmma_bf16(af0, bf1, acc[0][1]);
    acc[1][0] = wmma_bf16(af1, bf0, acc[1][0]);
    acc[1][1] = wmma_bf16(af1, bf1, acc[1][1]);
  }

#pragma unroll
  for (int i = 0; i < 2; ++i)
#pragma unroll
    for (int j = 0; j < 2; ++j)
#pragma unroll
      for (int r = 0; r < 8; ++r) {
        int gm = tileM + wrow + i * 16 + r + half8;
        int gn = tileN + wcol + j * 16 + l16;
        float v = acc[i][j][r];
        if (bias)  v += bias[gn];
        if (resid) v += resid[(size_t)gm * N + gn];
        if (relu)  v = fmaxf(v, 0.f);
        if (OutF)  OutF[(size_t)gm * N + gn] = v;
        if (OutB)  OutB[(size_t)gm * N + gn] = (bf16)v;
      }
}

// ---------------------------------------------------------------------------
// Im2col conv GEMM (bf16 in), templated on kernel size (1 or 3).
// In (8,Cin,32,32) NCHW -> Out (8,Cout,32,32) NCHW; Out = act(bn(conv+bias)).
// A tile: VGPR gather (bf16, no cvt). B tile: async copy.
// ---------------------------------------------------------------------------
template <int KS>
__global__ __launch_bounds__(256) void conv_bf16_kernel(
    const bf16* __restrict__ In, const bf16* __restrict__ Wt,
    const float* __restrict__ bias, const float* __restrict__ bng,
    const float* __restrict__ bnb, float* __restrict__ OutF,
    bf16* __restrict__ OutB, int Cin, int Cout, int relu)
{
  constexpr int PAD = KS / 2;
  constexpr int KSQ = KS * KS;
  __shared__ bf16 As[2][128][40];
  __shared__ bf16 Bs[2][64][40];
  constexpr unsigned ASZ = 128 * 40 * 2;
  constexpr unsigned BSZ = 64 * 40 * 2;
  const int tid  = threadIdx.x;
  const int lane = tid & 31, wave = tid >> 5;
  const int wrow = (wave >> 1) * 32, wcol = (wave & 1) * 32;
  const int tileM = blockIdx.y * 128, tileN = blockIdx.x * 64;
  const int half8 = (lane >> 4) << 3;
  const int l16   = lane & 15;
  const int Ktot  = Cin * KSQ;

  const int am = tid >> 1, ak0 = (tid & 1) << 4;   // A gather: row am, 16 k
  const int gm = tileM + am;
  const int bb = gm >> 10, hw = gm & 1023;
  const int hh = hw >> 5,  ww = hw & 31;

  const int brow = tid >> 2, bseg = tid & 3;       // B async: 256 copies
  const bf16* bgp = Wt + (size_t)min(tileN + brow, Cout - 1) * Ktot + bseg * 8;
  const unsigned boff = lds_addr(&Bs[0][0][0]) + brow * 80 + bseg * 16;

  bf16 fa[16];
  const bf16 z16 = (bf16)0.0f;

  auto gloadA = [&](int k0) {
#pragma unroll
    for (int kk = 0; kk < 16; ++kk) {
      int k = k0 + ak0 + kk;
      int cin, hi, wi;
      if (KS == 1) { cin = k; hi = hh; wi = ww; }
      else {
        cin = k / KSQ; int rem = k - cin * KSQ;
        int kh = rem / KS, kw = rem - kh * KS;
        hi = hh + kh - PAD; wi = ww + kw - PAD;
      }
      bool ok = ((unsigned)hi < 32u) && ((unsigned)wi < 32u);
      int hic = ok ? hi : 0, wic = ok ? wi : 0;
      bf16 v = In[(((size_t)bb * Cin + cin) << 10) + (hic << 5) + wic];
      fa[kk] = ok ? v : z16;
    }
  };
  auto lstoreA = [&](int buf) { pack16_store(fa, &As[buf][am][ak0]); };

  v8f acc[2][2] = {};
  const int KT = Ktot >> 5;
  gloadA(0);
  async_b128(boff, bgp);
  lstoreA(0);
  for (int kt = 0; kt < KT; ++kt) {
    wait_async0();
    __syncthreads();
    const int buf = kt & 1;
    if (kt + 1 < KT) {
      async_b128(boff + (buf ^ 1) * BSZ, bgp + ((kt + 1) << 5));
      gloadA((kt + 1) << 5);
    }
    v16bf af0 = load_frag(&As[buf][wrow + l16][0], half8);
    v16bf af1 = load_frag(&As[buf][wrow + 16 + l16][0], half8);
    v16bf bf0 = load_frag(&Bs[buf][wcol + l16][0], half8);
    v16bf bf1 = load_frag(&Bs[buf][wcol + 16 + l16][0], half8);
    acc[0][0] = wmma_bf16(af0, bf0, acc[0][0]);
    acc[0][1] = wmma_bf16(af0, bf1, acc[0][1]);
    acc[1][0] = wmma_bf16(af1, bf0, acc[1][0]);
    acc[1][1] = wmma_bf16(af1, bf1, acc[1][1]);
    if (kt + 1 < KT) lstoreA(buf ^ 1);
  }

  const float bnrs = rsqrtf(1.0f + 1e-5f);
#pragma unroll
  for (int i = 0; i < 2; ++i)
#pragma unroll
    for (int j = 0; j < 2; ++j)
#pragma unroll
      for (int r = 0; r < 8; ++r) {
        int gmo = tileM + wrow + i * 16 + r + half8;
        int gn  = tileN + wcol + j * 16 + l16;
        if (gn < Cout) {
          float v = acc[i][j][r];
          if (bias) v += bias[gn];
          if (bng)  v = v * (bng[gn] * bnrs) + bnb[gn];
          if (relu) v = fmaxf(v, 0.f);
          int bo = gmo >> 10, hwo = gmo & 1023;
          if (OutF) OutF[((size_t)bo * Cout + gn) * 1024 + hwo] = v;
          if (OutB) OutB[((size_t)bo * Cout + gn) * 1024 + hwo] = (bf16)v;
        }
      }
}

// ---------------------------------------------------------------------------
// Flash attention on bf16 qkv (S=1024, B=8, 3E=1152), 8 heads, dh=48 (pad 64).
// grid.x = 8 query blocks (128 rows), grid.y = b*8+h. Each wave: 16 q rows.
// K tile staged via async copies; V tile manually transposed into LDS.
// ---------------------------------------------------------------------------
__global__ __launch_bounds__(256) void attn_kernel(
    const bf16* __restrict__ qkv, bf16* __restrict__ o)
{
  __shared__ bf16 Kt[32][72];    // [key][dh 0..63]; tail 48..63 zeroed once
  __shared__ bf16 Vtt[48][40];   // [dh 0..47][key]  (wmma K-dim = key)
  __shared__ bf16 Pb[8][16][32]; // per-wave P re-layout buffer
  const int bh = blockIdx.y;
  const int b = bh >> 3, h = bh & 7;
  const int tid = threadIdx.x;
  const int lane = tid & 31, wave = tid >> 5;
  const int half8 = (lane >> 4) << 3;
  const int l16 = lane & 15;
  const int qbase = blockIdx.x * 128 + wave * 16;
  const float rs = 0.14433756729740643f;  // 1/sqrt(48)
  const unsigned ktBase = lds_addr(&Kt[0][0]);

  // zero Kt dh-tail once (async copies never touch it)
  if (tid < 64) {
    uint4 z = make_uint4(0u, 0u, 0u, 0u);
    *(uint4*)&Kt[tid >> 1][48 + ((tid & 1) << 3)] = z;
  }

  // Q fragments (16 rows x 64 dh, two K-steps), scale folded in
  v16bf qf[2];
  {
    const bf16* qrow = qkv + ((size_t)(qbase + l16) * 8 + b) * 1152 + h * 48;
#pragma unroll
    for (int st = 0; st < 2; ++st) {
      union { v16bf v; bf16 e[16]; } u;
#pragma unroll
      for (int e2 = 0; e2 < 16; ++e2) {
        int kd = st * 32 + half8 + (e2 & 7) + ((e2 >> 3) << 4);
        int kdc = kd < 48 ? kd : 0;
        float v = (float)qrow[kdc];
        u.e[e2] = (bf16)(kd < 48 ? v * rs : 0.f);
      }
      qf[st] = u.v;
    }
  }

  const int t2 = tid & 127;
  const int skey = t2 >> 2;            // V staging: key 0..31
  const int sd0  = (t2 & 3) << 4;      // 0,16,32,48

  float m8[8], l8[8];
  v8f oacc[3] = {};
#pragma unroll
  for (int r = 0; r < 8; ++r) { m8[r] = -3.0e38f; l8[r] = 0.f; }

  for (int kb = 0; kb < 32; ++kb) {
    if (tid < 128) {
      // async stage K: 32 keys x 96B = 192 x 16B copies over 128 lanes
#pragma unroll
      for (int rnd = 0; rnd < 2; ++rnd) {
        int t = tid + rnd * 128;
        if (t < 192) {
          int key = t / 6, seg = t - key * 6;
          const bf16* g = qkv +
              ((size_t)(kb * 32 + key) * 8 + b) * 1152 + 384 + h * 48 + seg * 8;
          async_b128(ktBase + key * 144 + seg * 16, g);
        }
      }
    } else if (sd0 != 48) {
      // manual transposed V stage
      const int s = kb * 32 + skey;
      const uint4* s4 =
          (const uint4*)(qkv + ((size_t)s * 8 + b) * 1152 + 768 + h * 48 + sd0);
      union { uint4 q[2]; bf16 e[16]; } u;
      u.q[0] = s4[0]; u.q[1] = s4[1];
#pragma unroll
      for (int i = 0; i < 16; ++i) Vtt[sd0 + i][skey] = u.e[i];
    }
    wait_async0();
    __syncthreads();

    // S = Q K^T : two 16x16 frags (keys 0-15, 16-31), two dh K-steps each
    v8f s0 = {}, s1 = {};
#pragma unroll
    for (int st = 0; st < 2; ++st) {
      v16bf b0 = load_frag(&Kt[l16][st * 32], half8);
      v16bf b1 = load_frag(&Kt[16 + l16][st * 32], half8);
      s0 = wmma_bf16(qf[st], b0, s0);
      s1 = wmma_bf16(qf[st], b1, s1);
    }

    // online softmax (rows in VGPR index, cols across 16 lanes)
    float p0[8], p1[8], alpha[8];
#pragma unroll
    for (int r = 0; r < 8; ++r) {
      float mx = fmaxf(s0[r], s1[r]);
      for (int off = 1; off < 16; off <<= 1)
        mx = fmaxf(mx, __shfl_xor(mx, off, 32));
      float mnew = fmaxf(m8[r], mx);
      float a = __expf(m8[r] - mnew);
      float e0 = __expf(s0[r] - mnew);
      float e1 = __expf(s1[r] - mnew);
      float rsum = e0 + e1;
      for (int off = 1; off < 16; off <<= 1)
        rsum += __shfl_xor(rsum, off, 32);
      m8[r] = mnew;
      l8[r] = l8[r] * a + rsum;
      alpha[r] = a; p0[r] = e0; p1[r] = e1;
    }
#pragma unroll
    for (int f = 0; f < 3; ++f)
#pragma unroll
      for (int r = 0; r < 8; ++r) oacc[f][r] *= alpha[r];

    // re-layout P (D-frag -> A-frag) through per-wave LDS
#pragma unroll
    for (int r = 0; r < 8; ++r) {
      int row = r + half8;
      Pb[wave][row][l16]      = (bf16)p0[r];
      Pb[wave][row][16 + l16] = (bf16)p1[r];
    }
    v16bf pa = load_frag(&Pb[wave][l16][0], half8);

    // O += P @ V  (three dh column frags: 0-15, 16-31, 32-47)
#pragma unroll
    for (int f = 0; f < 3; ++f) {
      v16bf vf = load_frag(&Vtt[f * 16 + l16][0], half8);
      oacc[f] = wmma_bf16(pa, vf, oacc[f]);
    }
    __syncthreads();
  }

  // normalize + store o[s, b, h*48+d] as bf16 (d = f*16 + l16 < 48 always)
#pragma unroll
  for (int f = 0; f < 3; ++f)
#pragma unroll
    for (int r = 0; r < 8; ++r) {
      int s = qbase + r + half8;
      o[((size_t)s * 8 + b) * 384 + h * 48 + f * 16 + l16] =
          (bf16)(oacc[f][r] / l8[r]);
    }
}

// ---------------------------------------------------------------------------
// LayerNorm over rows of 384 (8192 rows); optional bf16 mirror output
// ---------------------------------------------------------------------------
__global__ __launch_bounds__(256) void ln_kernel(
    const float* __restrict__ x, const float* __restrict__ g,
    const float* __restrict__ beta, float* __restrict__ outF,
    bf16* __restrict__ outB)
{
  int row = blockIdx.x * 8 + (threadIdx.x >> 5);
  int lane = threadIdx.x & 31;
  const float* xr = x + (size_t)row * 384;
  float s = 0.f;
  for (int i = lane; i < 384; i += 32) s += xr[i];
  for (int off = 1; off < 32; off <<= 1) s += __shfl_xor(s, off, 32);
  float mean = s * (1.f / 384.f);
  float v = 0.f;
  for (int i = lane; i < 384; i += 32) { float d = xr[i] - mean; v += d * d; }
  for (int off = 1; off < 32; off <<= 1) v += __shfl_xor(v, off, 32);
  float inv = rsqrtf(v * (1.f / 384.f) + 1e-5f);
  for (int i = lane; i < 384; i += 32) {
    float y = (xr[i] - mean) * inv * g[i] + beta[i];
    outF[(size_t)row * 384 + i] = y;
    if (outB) outB[(size_t)row * 384 + i] = (bf16)y;
  }
}

// (S,B,E) f32 row-major -> (B,E,H,W) bf16
__global__ __launch_bounds__(256) void to_nchw_kernel(
    const float* __restrict__ src2, bf16* __restrict__ mm)
{
  int idx = blockIdx.x * 256 + threadIdx.x;  // over 8*384*1024
  int b = idx / (384 * 1024); int rem = idx - b * 384 * 1024;
  int e = rem >> 10; int hw = rem & 1023;
  mm[idx] = (bf16)src2[((size_t)hw * 8 + b) * 384 + e];
}

// involution apply: dyn[b,g*16+c,h,w] = sum_t wgt[b,g*9+t,h,w]*ypad[b,c,h+i,w+j]
__global__ __launch_bounds__(256) void involution_kernel(
    const float* __restrict__ y, const float* __restrict__ wgt,
    float* __restrict__ dynOut, bf16* __restrict__ cat, int mod)
{
  int idx = blockIdx.x * 256 + threadIdx.x;  // 8*128*1024
  int b = idx >> 17; int c = (idx >> 10) & 127; int hw = idx & 1023;
  int hh = hw >> 5, ww = hw & 31;
  int g = c >> 4;
  float acc = 0.f;
#pragma unroll
  for (int t = 0; t < 9; ++t) {
    int i = t / 3, j = t - i * 3;
    int hi = hh + i - 1, wi = ww + j - 1;
    bool ok = ((unsigned)hi < 32u) && ((unsigned)wi < 32u);
    int hic = ok ? hi : 0, wic = ok ? wi : 0;
    float wv = wgt[((size_t)b * 72 + g * 9 + t) * 1024 + hw];
    float yv = y[((size_t)b * 128 + c) * 1024 + (hic << 5) + wic];
    acc += ok ? wv * yv : 0.f;
  }
  dynOut[idx] = acc;
  cat[((size_t)b * 384 + mod * 128 + c) * 1024 + hw] = (bf16)acc;
}

// ---------------------------------------------------------------------------
extern "C" void kernel_launch(void* const* d_in, const int* in_sizes, int n_in,
                              void* d_out, int out_size, void* d_ws, size_t ws_size,
                              hipStream_t stream)
{
  const float* src       = (const float*)d_in[0];
  const float* depth     = (const float*)d_in[1];
  const float* sal       = (const float*)d_in[2];
  const float* contour   = (const float*)d_in[3];
  const float* ln1_g     = (const float*)d_in[4];
  const float* ln1_b     = (const float*)d_in[5];
  const float* attn_in_w = (const float*)d_in[6];
  const float* attn_in_b = (const float*)d_in[7];
  const float* attn_out_w= (const float*)d_in[8];
  const float* attn_out_b= (const float*)d_in[9];
  const float* ln2_g     = (const float*)d_in[10];
  const float* ln2_b     = (const float*)d_in[11];
  const float* ffn_w1    = (const float*)d_in[12];
  const float* ffn_b1    = (const float*)d_in[13];
  const float* ffn_w2    = (const float*)d_in[14];
  const float* ffn_b2    = (const float*)d_in[15];
  const float* br_w      = (const float*)d_in[16];
  const float* br_b      = (const float*)d_in[17];
  const float* br_g      = (const float*)d_in[18];
  const float* br_beta   = (const float*)d_in[19];
  const float* c1_w      = (const float*)d_in[20];
  const float* c1_g      = (const float*)d_in[21];
  const float* c1_b      = (const float*)d_in[22];
  const float* c2_w      = (const float*)d_in[23];
  const float* c2_b      = (const float*)d_in[24];
  const float* fuse_w    = (const float*)d_in[25];
  const float* fuse_b    = (const float*)d_in[26];
  const float* fuse_g    = (const float*)d_in[27];
  const float* fuse_beta = (const float*)d_in[28];
  (void)in_sizes; (void)n_in; (void)out_size; (void)ws_size;

  // ---- workspace carve-up (all sizes are multiples of 256B) ----
  char* p = (char*)d_ws;
  auto allocF = [&](size_t n) { float* r = (float*)p; p += n * 4; return r; };
  auto allocB = [&](size_t n) { bf16*  r = (bf16*)p;  p += n * 2; return r; };
  // f32 intermediates
  float* r1    = allocF(3145728);
  float* src1  = allocF(3145728);
  float* src2  = allocF(3145728);
  float* ybuf  = allocF(3145728);   // 3 x (8,128,32,32)
  float* wgtb  = allocF(1769472);   // 3 x (8,72,32,32)
  // bf16 activations
  bf16* src_bf  = allocB(3145728);
  bf16* qkv_bf  = allocB(9437184);
  bf16* o_bf    = allocB(3145728);
  bf16* src1_bf = allocB(3145728);
  bf16* hid_bf  = allocB(12582912);
  bf16* mm_bf   = allocB(3145728);
  bf16* c1h_bf  = allocB(786432);
  bf16* cat_bf  = allocB(3145728);
  bf16* x_bf    = allocB(3145728);  // depth/contour/sal, 3 x 1,048,576
  // bf16 weights
  bf16* wq_bf   = allocB(442368);
  bf16* wo_bf   = allocB(147456);
  bf16* w1_bf   = allocB(589824);
  bf16* w2_bf   = allocB(589824);
  bf16* brw_bf  = allocB(1327104);
  bf16* c1w_bf  = allocB(12288);
  bf16* c2w_bf  = allocB(6912);
  bf16* fusew_bf= allocB(1327104);
  float* outF = (float*)d_out;

  dim3 blk(256);
  auto cvt = [&](const float* in, bf16* out, int n) {
    cvt_bf16_kernel<<<dim3((n / 8 + 255) / 256), blk, 0, stream>>>(in, out, n);
  };

  // ---- one-time bf16 conversions ----
  cvt(src, src_bf, 3145728);
  cvt(attn_in_w, wq_bf, 442368);
  cvt(attn_out_w, wo_bf, 147456);
  cvt(ffn_w1, w1_bf, 589824);
  cvt(ffn_w2, w2_bf, 589824);
  cvt(br_w, brw_bf, 1327104);
  cvt(c1_w, c1w_bf, 12288);
  cvt(c2_w, c2w_bf, 6912);
  cvt(fuse_w, fusew_bf, 1327104);
  cvt(depth,   x_bf,           1048576);  // xs[0]
  cvt(contour, x_bf + 1048576, 1048576);  // xs[1]
  cvt(sal,     x_bf + 2097152, 1048576);  // xs[2]

  // 1) QKV: (8192x384) @ (1152x384)^T -> bf16
  gemm_bf16_kernel<<<dim3(18, 64), blk, 0, stream>>>(
      src_bf, wq_bf, attn_in_b, nullptr, nullptr, qkv_bf, 8192, 1152, 384, 0);
  // 2) attention -> bf16
  attn_kernel<<<dim3(8, 64), blk, 0, stream>>>(qkv_bf, o_bf);
  // 3) out proj + residual(src, f32) -> f32
  gemm_bf16_kernel<<<dim3(6, 64), blk, 0, stream>>>(
      o_bf, wo_bf, attn_out_b, src, r1, nullptr, 8192, 384, 384, 0);
  // 4) LN1 -> f32 + bf16
  ln_kernel<<<dim3(1024), blk, 0, stream>>>(r1, ln1_g, ln1_b, src1, src1_bf);
  // 5) FFN1 (relu) -> bf16
  gemm_bf16_kernel<<<dim3(24, 64), blk, 0, stream>>>(
      src1_bf, w1_bf, ffn_b1, nullptr, nullptr, hid_bf, 8192, 1536, 384, 1);
  // 6) FFN2 + residual(src1) -> f32
  gemm_bf16_kernel<<<dim3(6, 64), blk, 0, stream>>>(
      hid_bf, w2_bf, ffn_b2, src1, r1, nullptr, 8192, 384, 1536, 0);
  // 7) LN2 -> f32
  ln_kernel<<<dim3(1024), blk, 0, stream>>>(r1, ln2_g, ln2_b, src2, nullptr);
  // 8) reshape to (B,E,H,W) bf16
  to_nchw_kernel<<<dim3(12288), blk, 0, stream>>>(src2, mm_bf);

  for (int m = 0; m < 3; ++m) {
    // 9) branch: 3x3 conv 384->128 + bias + bn + relu -> f32
    conv_bf16_kernel<3><<<dim3(2, 64), blk, 0, stream>>>(
        mm_bf, brw_bf + (size_t)m * 442368, br_b + m * 128, br_g + m * 128,
        br_beta + m * 128, ybuf + (size_t)m * 1048576, nullptr, 384, 128, 1);
    // 10) dyn c1: 1x1 conv 128->32 + bn + relu (no bias) -> bf16
    conv_bf16_kernel<1><<<dim3(1, 64), blk, 0, stream>>>(
        x_bf + (size_t)m * 1048576, c1w_bf + (size_t)m * 4096, nullptr,
        c1_g + m * 32, c1_b + m * 32, nullptr, c1h_bf + (size_t)m * 262144,
        128, 32, 1);
    // 11) dyn c2: 1x1 conv 32->72 + bias -> f32
    conv_bf16_kernel<1><<<dim3(2, 64), blk, 0, stream>>>(
        c1h_bf + (size_t)m * 262144, c2w_bf + (size_t)m * 2304, c2_b + m * 72,
        nullptr, nullptr, wgtb + (size_t)m * 589824, nullptr, 32, 72, 0);
  }
  // 12) involution apply -> dyn outputs (f32) + cat buffer (bf16)
  for (int m = 0; m < 3; ++m)
    involution_kernel<<<dim3(4096), blk, 0, stream>>>(
        ybuf + (size_t)m * 1048576, wgtb + (size_t)m * 589824,
        outF + (size_t)m * 1048576, cat_bf, m);
  // 13) fuse: 3x3 conv 384->384 + bias + bn + relu -> f32
  conv_bf16_kernel<3><<<dim3(6, 64), blk, 0, stream>>>(
      cat_bf, fusew_bf, fuse_b, fuse_g, fuse_beta, outF + 3145728, nullptr,
      384, 384, 1);
}
